// Net_40226663694756
// MI455X (gfx1250) — matrix operand order repaired and checked by reading, not compiled, not run on hardware
//
#include <hip/hip_runtime.h>
#include <hip/hip_bf16.h>
#include <math.h>

#define N_NODES 100000
#define E_EDGES 1600000
#define ETOT    (E_EDGES + N_NODES)
#define NEG_SLOPE 0.2f

typedef __attribute__((ext_vector_type(16))) __bf16 v16bf;
typedef __attribute__((ext_vector_type(8)))  float  v8f;

union ABf16 { v16bf v; unsigned short u[16]; uint4 q[2]; };

__device__ __forceinline__ unsigned short f32_to_bf16(float f) {
  unsigned int u = __float_as_uint(f);
  u += 0x7FFFu + ((u >> 16) & 1u);
  return (unsigned short)(u >> 16);
}
__device__ __forceinline__ unsigned int fkey(float f) {
  unsigned int u = __float_as_uint(f);
  return (u & 0x80000000u) ? ~u : (u | 0x80000000u);
}
__device__ __forceinline__ float funkey(unsigned int k) {
  unsigned int u = (k & 0x80000000u) ? (k & 0x7FFFFFFFu) : ~k;
  return __uint_as_float(u);
}
__device__ __forceinline__ float lrelu(float v) { return v > 0.f ? v : NEG_SLOPE * v; }

// ---------------------------------------------------------------------------
// Pre-conversion kernels (run once per call; keeps WMMA hot loop pure)
// ---------------------------------------------------------------------------
__global__ void cvt_f32_bf16(const float* __restrict__ in,
                             unsigned short* __restrict__ out, int n) {
  int t = blockIdx.x * blockDim.x + threadIdx.x;
  if (t < n) out[t] = f32_to_bf16(in[t]);
}

// W[K,NC] f32 row-major  ->  Wt[NC,K] bf16 (transposed)
__global__ void cvt_transpose_bf16(const float* __restrict__ W,
                                   unsigned short* __restrict__ Wt, int K, int NC) {
  int t = blockIdx.x * blockDim.x + threadIdx.x;
  if (t >= K * NC) return;
  const int k = t / NC, c = t % NC;
  Wt[(size_t)c * K + k] = f32_to_bf16(W[t]);
}

// fused ELU + bf16 convert (layer-2 GEMM input)
__global__ void elu_cvt_bf16(const float* __restrict__ in,
                             unsigned short* __restrict__ out, int n) {
  int t = blockIdx.x * blockDim.x + threadIdx.x;
  if (t < n) {
    float v = in[t];
    v = v > 0.f ? v : (__expf(v) - 1.f);
    out[t] = f32_to_bf16(v);
  }
}

// ---------------------------------------------------------------------------
// H[M,NC] = Xb[M,K] @ Wt^T   (bf16 WMMA, f32 accum)
// One wave computes a (16*MT) x (16*NT) tile; B tiles reused MT times.
// A layout (16x32 bf16, ISA 7.12.2): lane%16 = row; lane/16 selects K-halves:
//   elems 0..7  -> K = k0 + (lane/16)*8 + e          (contiguous, 16B)
//   elems 8..15 -> K = k0 + 16 + (lane/16)*8 + (e-8) (contiguous, 16B)
// B layout (32x16 bf16): lane%16 = col; same K mapping as A
//   -> with Wt[NC,K] this is two 16B contiguous chunks per lane.
// C/D (16x16 f32): vgpr r, lanes 0-15 -> M=r, lanes 16-31 -> M=r+8.
// ---------------------------------------------------------------------------
template <int MT, int NT>
__global__ __launch_bounds__(128) void gemm_bf16_wmma(
    const unsigned short* __restrict__ Xb, const unsigned short* __restrict__ Wt,
    float* __restrict__ H, int M, int K) {
  const int NC   = NT * 16;
  const int wave = threadIdx.x >> 5;
  const int lane = threadIdx.x & 31;
  const int row0 = (blockIdx.x * (blockDim.x >> 5) + wave) * (16 * MT);
  if (row0 >= M) return;

  const int rlo   = lane & 15;
  const int koff  = (lane >> 4) * 8;
  const int col   = lane & 15;
  const int khalf = (lane >> 4) * 16;

  v8f acc[MT][NT];
#pragma unroll
  for (int mt = 0; mt < MT; ++mt)
#pragma unroll
    for (int t = 0; t < NT; ++t)
#pragma unroll
      for (int i = 0; i < 8; ++i) acc[mt][t][i] = 0.f;

  for (int k0 = 0; k0 < K; k0 += 32) {
    ABf16 a[MT];
#pragma unroll
    for (int mt = 0; mt < MT; ++mt) {
      const unsigned short* xp =
          Xb + (size_t)(row0 + mt * 16 + rlo) * K + k0 + koff;
      a[mt].q[0] = *(const uint4*)(xp);
      a[mt].q[1] = *(const uint4*)(xp + 16);
    }
#pragma unroll
    for (int t = 0; t < NT; ++t) {
      ABf16 b;
      const unsigned short* wp = Wt + (size_t)(t * 16 + col) * K + k0 + koff;
      b.q[0] = *(const uint4*)(wp);
      b.q[1] = *(const uint4*)(wp + 16);
#pragma unroll
      for (int mt = 0; mt < MT; ++mt)
        acc[mt][t] = __builtin_amdgcn_wmma_f32_16x16x32_bf16(
            false, a[mt].v, false, b.v, (short)0, acc[mt][t], false, false);
    }
  }

#pragma unroll
  for (int mt = 0; mt < MT; ++mt) {
    const int rbase = row0 + mt * 16 + (lane >> 4) * 8;
#pragma unroll
    for (int t = 0; t < NT; ++t) {
      const int c = t * 16 + col;
#pragma unroll
      for (int rr = 0; rr < 8; ++rr)
        H[(size_t)(rbase + rr) * NC + c] = acc[mt][t][rr];
    }
  }
}

// ---------------------------------------------------------------------------
// Per-node attention logits: a[n,h] = sum_c H[n,h*ch+c] * att[h*ch+c]
// ---------------------------------------------------------------------------
__global__ void node_att(const float* __restrict__ H, const float* __restrict__ att_s,
                         const float* __restrict__ att_d, float* __restrict__ as_,
                         float* __restrict__ ad_, int n, int heads, int ch) {
  int tid = blockIdx.x * blockDim.x + threadIdx.x;
  if (tid >= n * heads) return;
  const int node = tid / heads, h = tid % heads;
  const float* hp = H + (size_t)node * heads * ch + h * ch;
  const float* sp = att_s + h * ch;
  const float* dp = att_d + h * ch;
  float s = 0.f, d = 0.f;
  for (int c = 0; c < ch; ++c) { const float v = hp[c]; s += v * sp[c]; d += v * dp[c]; }
  as_[tid] = s;
  ad_[tid] = d;
}

__device__ __forceinline__ void edge_sd(const int* __restrict__ src,
                                        const int* __restrict__ dst,
                                        int e, int ne, int& s, int& d) {
  if (e < ne) { s = src[e]; d = dst[e]; } else { s = e - ne; d = e - ne; }
}

__global__ void edge_max(const int* __restrict__ src, const int* __restrict__ dst,
                         const float* __restrict__ as_, const float* __restrict__ ad_,
                         unsigned int* __restrict__ mk, int etot, int ne, int heads) {
  int e = blockIdx.x * blockDim.x + threadIdx.x;
  if (e >= etot) return;
  int s, d; edge_sd(src, dst, e, ne, s, d);
  for (int h = 0; h < heads; ++h) {
    float v = lrelu(as_[s * heads + h] + ad_[d * heads + h]);
    atomicMax(&mk[d * heads + h], fkey(v));
  }
}

__global__ void edge_expsum(const int* __restrict__ src, const int* __restrict__ dst,
                            const float* __restrict__ as_, const float* __restrict__ ad_,
                            const unsigned int* __restrict__ mk, float* __restrict__ ee,
                            float* __restrict__ denom, int etot, int ne, int heads) {
  int e = blockIdx.x * blockDim.x + threadIdx.x;
  if (e >= etot) return;
  int s, d; edge_sd(src, dst, e, ne, s, d);
  for (int h = 0; h < heads; ++h) {
    float v  = lrelu(as_[s * heads + h] + ad_[d * heads + h]);
    float ex = __expf(v - funkey(mk[d * heads + h]));
    ee[(size_t)e * heads + h] = ex;
    atomicAdd(&denom[d * heads + h], ex);
  }
}

// thread = (edge, float4 chunk): wave reads one coalesced row slice of H[src]
__global__ void edge_aggregate(const int* __restrict__ src, const int* __restrict__ dst,
                               const float* __restrict__ ee, const float* __restrict__ denom,
                               const float* __restrict__ Hs, float* __restrict__ out,
                               int etot, int ne, int heads, int nc, int chunksPerHead) {
  const int chunks = nc >> 2;
  int tid = blockIdx.x * blockDim.x + threadIdx.x;
  int e = tid / chunks;
  int c4 = tid % chunks;
  if (e >= etot) return;
  int s, d; edge_sd(src, dst, e, ne, s, d);
  const int h = c4 / chunksPerHead;
  const float alpha = ee[(size_t)e * heads + h] / (denom[d * heads + h] + 1e-16f);
  const float4 hv = *(const float4*)(Hs + (size_t)s * nc + c4 * 4);
  float* op = out + (size_t)d * nc + c4 * 4;
  atomicAdd(op + 0, alpha * hv.x);
  atomicAdd(op + 1, alpha * hv.y);
  atomicAdd(op + 2, alpha * hv.z);
  atomicAdd(op + 3, alpha * hv.w);
}

__global__ void fill_u32(unsigned int* __restrict__ p, unsigned int v, int n) {
  int t = blockIdx.x * blockDim.x + threadIdx.x;
  if (t < n) p[t] = v;
}
__global__ void fill_f32(float* __restrict__ p, float v, int n) {
  int t = blockIdx.x * blockDim.x + threadIdx.x;
  if (t < n) p[t] = v;
}
__global__ void init_rows_bias(float* __restrict__ out, const float* __restrict__ bias,
                               int n, int nc) {
  int t = blockIdx.x * blockDim.x + threadIdx.x;
  if (t < n * nc) out[t] = bias[t % nc];
}

// log_softmax over 64 classes: one wave per row, 2 elements/lane, wave32 shuffles
__global__ void log_softmax64(float* __restrict__ out, int n) {
  const int wave = threadIdx.x >> 5, lane = threadIdx.x & 31;
  const int row = blockIdx.x * (blockDim.x >> 5) + wave;
  if (row >= n) return;
  float* p = out + (size_t)row * 64;
  const float v0 = p[lane], v1 = p[lane + 32];
  float m = fmaxf(v0, v1);
#pragma unroll
  for (int o = 16; o >= 1; o >>= 1) m = fmaxf(m, __shfl_xor(m, o, 32));
  float s = __expf(v0 - m) + __expf(v1 - m);
#pragma unroll
  for (int o = 16; o >= 1; o >>= 1) s += __shfl_xor(s, o, 32);
  const float lse = m + __logf(s);
  p[lane] = v0 - lse;
  p[lane + 32] = v1 - lse;
}

extern "C" void kernel_launch(void* const* d_in, const int* in_sizes, int n_in,
                              void* d_out, int out_size, void* d_ws, size_t ws_size,
                              hipStream_t stream) {
  (void)in_sizes; (void)n_in; (void)out_size; (void)ws_size;
  const float* x        = (const float*)d_in[0];
  const int*   eidx     = (const int*)d_in[1];
  const float* W1       = (const float*)d_in[2];
  const float* att_src1 = (const float*)d_in[3];
  const float* att_dst1 = (const float*)d_in[4];
  const float* b1       = (const float*)d_in[5];
  const float* W2       = (const float*)d_in[6];
  const float* att_src2 = (const float*)d_in[7];
  const float* att_dst2 = (const float*)d_in[8];
  const float* b2       = (const float*)d_in[9];
  float* out = (float*)d_out;

  const int* src = eidx;
  const int* dst = eidx + E_EDGES;

  // ---- workspace layout ----
  float* w = (float*)d_ws;
  float*        H1  = w;                 w += (size_t)N_NODES * 128;
  float*        AS1 = w;                 w += (size_t)N_NODES * 8;
  float*        AD1 = w;                 w += (size_t)N_NODES * 8;
  unsigned int* MK1 = (unsigned int*)w;  w += (size_t)N_NODES * 8;
  float*        DN1 = w;                 w += (size_t)N_NODES * 8;
  float*        EE1 = w;                 w += (size_t)ETOT * 8;
  float*        X2  = w;                 w += (size_t)N_NODES * 128;
  float*        H2  = w;                 w += (size_t)N_NODES * 64;
  float*        AS2 = w;                 w += (size_t)N_NODES;
  float*        AD2 = w;                 w += (size_t)N_NODES;
  unsigned int* MK2 = (unsigned int*)w;  w += (size_t)N_NODES;
  float*        DN2 = w;                 w += (size_t)N_NODES;
  float*        EE2 = w;                 w += (size_t)ETOT;
  unsigned short* Xb  = (unsigned short*)w;  // [N,256] bf16
  unsigned short* W1t = Xb + (size_t)N_NODES * 256;   // [128,256] bf16 transposed
  unsigned short* W2t = W1t + 128 * 256;              // [64,128]  bf16 transposed
  unsigned short* X2b = W2t + 64 * 128;               // [N,128]   bf16

  const unsigned int KNEG_INF = ~0xFF800000u;  // fkey(-inf)
  const int TB = 256;

  // ---- one-time (per call) conversions: keep WMMA loops pure bf16 loads ----
  cvt_f32_bf16<<<(N_NODES * 256 + TB - 1) / TB, TB, 0, stream>>>(x, Xb, N_NODES * 256);
  cvt_transpose_bf16<<<(256 * 128 + TB - 1) / TB, TB, 0, stream>>>(W1, W1t, 256, 128);
  cvt_transpose_bf16<<<(128 * 64 + TB - 1) / TB, TB, 0, stream>>>(W2, W2t, 128, 64);

  // ---------------- Layer 1 (heads=8, ch=16, concat) ----------------
  gemm_bf16_wmma<2, 8><<<(N_NODES + 127) / 128, 128, 0, stream>>>(Xb, W1t, H1, N_NODES, 256);
  node_att<<<(N_NODES * 8 + TB - 1) / TB, TB, 0, stream>>>(H1, att_src1, att_dst1,
                                                           AS1, AD1, N_NODES, 8, 16);
  fill_u32<<<(N_NODES * 8 + TB - 1) / TB, TB, 0, stream>>>(MK1, KNEG_INF, N_NODES * 8);
  fill_f32<<<(N_NODES * 8 + TB - 1) / TB, TB, 0, stream>>>(DN1, 0.f, N_NODES * 8);
  edge_max<<<(ETOT + TB - 1) / TB, TB, 0, stream>>>(src, dst, AS1, AD1, MK1, ETOT, E_EDGES, 8);
  edge_expsum<<<(ETOT + TB - 1) / TB, TB, 0, stream>>>(src, dst, AS1, AD1, MK1, EE1, DN1,
                                                       ETOT, E_EDGES, 8);
  init_rows_bias<<<(N_NODES * 128 + TB - 1) / TB, TB, 0, stream>>>(X2, b1, N_NODES, 128);
  {
    const long long work = (long long)ETOT * 32;
    edge_aggregate<<<(int)((work + TB - 1) / TB), TB, 0, stream>>>(
        src, dst, EE1, DN1, H1, X2, ETOT, E_EDGES, 8, 128, 4);
  }
  elu_cvt_bf16<<<(N_NODES * 128 + TB - 1) / TB, TB, 0, stream>>>(X2, X2b, N_NODES * 128);

  // ---------------- Layer 2 (heads=1, ch=64, mean==identity) ----------------
  gemm_bf16_wmma<2, 4><<<(N_NODES + 127) / 128, 128, 0, stream>>>(X2b, W2t, H2, N_NODES, 128);
  node_att<<<(N_NODES + TB - 1) / TB, TB, 0, stream>>>(H2, att_src2, att_dst2,
                                                       AS2, AD2, N_NODES, 1, 64);
  fill_u32<<<(N_NODES + TB - 1) / TB, TB, 0, stream>>>(MK2, KNEG_INF, N_NODES);
  fill_f32<<<(N_NODES + TB - 1) / TB, TB, 0, stream>>>(DN2, 0.f, N_NODES);
  edge_max<<<(ETOT + TB - 1) / TB, TB, 0, stream>>>(src, dst, AS2, AD2, MK2, ETOT, E_EDGES, 1);
  edge_expsum<<<(ETOT + TB - 1) / TB, TB, 0, stream>>>(src, dst, AS2, AD2, MK2, EE2, DN2,
                                                       ETOT, E_EDGES, 1);
  init_rows_bias<<<(N_NODES * 64 + TB - 1) / TB, TB, 0, stream>>>(out, b2, N_NODES, 64);
  {
    const long long work = (long long)ETOT * 16;
    edge_aggregate<<<(int)((work + TB - 1) / TB), TB, 0, stream>>>(
        src, dst, EE2, DN2, H2, out, ETOT, E_EDGES, 1, 64, 64);
  }
  log_softmax64<<<(N_NODES + 3) / 4, 128, 0, stream>>>(out, N_NODES);
}
